// Ng_encoder_6579889898154
// MI455X (gfx1250) — matrix operand order repaired and checked by reading, not compiled, not run on hardware
//
#include <hip/hip_runtime.h>
#include <hip/hip_bf16.h>
#include <math.h>

// ---------------------------------------------------------------------------
// MI455X / gfx1250, wave32. D is fixed at 64 (2 floats per lane).
// ---------------------------------------------------------------------------

typedef __attribute__((ext_vector_type(16))) __bf16 v16bf;
typedef __attribute__((ext_vector_type(8)))  float  v8f;

#define GBLOCKS 240
#define GPARTS  (GBLOCKS * 8)   // waves in the global-attention pass

__device__ __forceinline__ float wave_reduce_sum(float v) {
#pragma unroll
  for (int m = 16; m >= 1; m >>= 1) v += __shfl_xor(v, m, 32);
  return v;
}

// ---------------------------------------------------------------------------
// Kernel A: neighbor attention + ELU.  One wave per node; lane l owns dims
// (2l, 2l+1).  Gathers hit L2 (tables are ~10 MB << 192 MB L2).
// ---------------------------------------------------------------------------
template <int S>
__global__ void nei_att_kernel(const float* __restrict__ href,   // N x 64
                               const float* __restrict__ hnei,   // M x 64
                               const int*   __restrict__ nei,    // N x S
                               const float* __restrict__ attw,   // 2*64
                               float* __restrict__ eout,         // N x 64
                               int nRows) {
  const int wave = threadIdx.x >> 5;
  const int lane = threadIdx.x & 31;
  const int n = blockIdx.x * (blockDim.x >> 5) + wave;
  if (n >= nRows) return;                      // wave-uniform

  const float2 aref = *(const float2*)(attw + 2 * lane);
  const float2 anei = *(const float2*)(attw + 64 + 2 * lane);
  const float2 h    = *(const float2*)(href + (size_t)n * 64 + 2 * lane);
  const float hdot  = wave_reduce_sum(aref.x * h.x + aref.y * h.y);

  float2 nv[S];
  float  sc[S];
#pragma unroll
  for (int s = 0; s < S; ++s) {
    const int idx = nei[(size_t)n * S + s];
    nv[s] = *(const float2*)(hnei + (size_t)idx * 64 + 2 * lane);
    const float d = wave_reduce_sum(anei.x * nv[s].x + anei.y * nv[s].y);
    const float t = hdot + d;
    sc[s] = t > 0.f ? t : 0.01f * t;           // leaky_relu(0.01)
  }
  float m = sc[0];
#pragma unroll
  for (int s = 1; s < S; ++s) m = fmaxf(m, sc[s]);
  float sum = 0.f;
#pragma unroll
  for (int s = 0; s < S; ++s) { sc[s] = __expf(sc[s] - m); sum += sc[s]; }
  const float inv = 1.f / sum;
  float2 acc = {0.f, 0.f};
#pragma unroll
  for (int s = 0; s < S; ++s) {
    const float w = sc[s] * inv;
    acc.x += w * nv[s].x;
    acc.y += w * nv[s].y;
  }
  acc.x = acc.x > 0.f ? acc.x : (__expf(acc.x) - 1.f);   // ELU
  acc.y = acc.y > 0.f ? acc.y : (__expf(acc.y) - 1.f);
  *(float2*)(eout + (size_t)n * 64 + 2 * lane) = acc;
}

// Fallback for unexpected S (two-pass, re-gathers instead of caching)
__global__ void nei_att_dyn_kernel(const float* __restrict__ href,
                                   const float* __restrict__ hnei,
                                   const int*   __restrict__ nei,
                                   const float* __restrict__ attw,
                                   float* __restrict__ eout,
                                   int nRows, int S) {
  const int wave = threadIdx.x >> 5;
  const int lane = threadIdx.x & 31;
  const int n = blockIdx.x * (blockDim.x >> 5) + wave;
  if (n >= nRows) return;
  const float2 aref = *(const float2*)(attw + 2 * lane);
  const float2 anei = *(const float2*)(attw + 64 + 2 * lane);
  const float2 h    = *(const float2*)(href + (size_t)n * 64 + 2 * lane);
  const float hdot  = wave_reduce_sum(aref.x * h.x + aref.y * h.y);
  float m = -INFINITY;
  for (int s = 0; s < S; ++s) {
    const int idx = nei[(size_t)n * S + s];
    const float2 v = *(const float2*)(hnei + (size_t)idx * 64 + 2 * lane);
    float t = hdot + wave_reduce_sum(anei.x * v.x + anei.y * v.y);
    t = t > 0.f ? t : 0.01f * t;
    m = fmaxf(m, t);
  }
  float sum = 0.f; float2 acc = {0.f, 0.f};
  for (int s = 0; s < S; ++s) {
    const int idx = nei[(size_t)n * S + s];
    const float2 v = *(const float2*)(hnei + (size_t)idx * 64 + 2 * lane);
    float t = hdot + wave_reduce_sum(anei.x * v.x + anei.y * v.y);
    t = t > 0.f ? t : 0.01f * t;
    const float w = __expf(t - m);
    sum += w; acc.x += w * v.x; acc.y += w * v.y;
  }
  acc.x /= sum; acc.y /= sum;
  acc.x = acc.x > 0.f ? acc.x : (__expf(acc.x) - 1.f);
  acc.y = acc.y > 0.f ? acc.y : (__expf(acc.y) - 1.f);
  *(float2*)(eout + (size_t)n * 64 + 2 * lane) = acc;
}

// ---------------------------------------------------------------------------
// Kernel B: beta partial via WMMA.  sum_n fus . tanh(e @ fc_w^T + b).
// One wave per 16-row tile of e; 4 col-tiles x 2 K-steps of
// v_wmma_f32_16x16x32_bf16.  Requires nRows % 16 == 0 (60000 = 16*3750).
// ---------------------------------------------------------------------------
__global__ void beta_partial_kernel(const float* __restrict__ e,     // N x 64
                                    const float* __restrict__ fcw,   // 64 x 64
                                    const float* __restrict__ fcb,   // 64
                                    const float* __restrict__ fus,   // 64
                                    float* __restrict__ beta_accum,  // 1 float
                                    int nRows) {
  const int wave = threadIdx.x >> 5;
  const int lane = threadIdx.x & 31;
  const int tile = blockIdx.x * (blockDim.x >> 5) + wave;
  const int r0 = tile * 16;
  if (r0 >= nRows) return;                     // wave-uniform
  const bool hi = lane >= 16;
  const int  ml = lane & 15;

  // A: rows r0..r0+15, K pairs per ISA 16-bit A layout (lane half => +8)
  v16bf A[2];
  const float* erow = e + (size_t)(r0 + ml) * 64;
#pragma unroll
  for (int ks = 0; ks < 2; ++ks) {
#pragma unroll
    for (int p = 0; p < 8; ++p) {
      const int k = ks * 32 + (p < 4 ? 2 * p : 16 + 2 * (p - 4)) + (hi ? 8 : 0);
      const float2 v = *(const float2*)(erow + k);
      A[ks][2 * p]     = (__bf16)v.x;
      A[ks][2 * p + 1] = (__bf16)v.y;
    }
  }

  float partial = 0.f;
#pragma unroll
  for (int jt = 0; jt < 4; ++jt) {
    const int j = jt * 16 + ml;                // column owned by this lane
    v8f acc = {};
#pragma unroll
    for (int ks = 0; ks < 2; ++ks) {
      // B[k][j] = fc_w[j*64 + k]; K pairs per ISA 16-bit layout (lane half => +8), same as A
      v16bf B;
      const float* wrow = fcw + (size_t)j * 64 + ks * 32 + (hi ? 8 : 0);
#pragma unroll
      for (int p = 0; p < 8; ++p) {
        const float2 v = *(const float2*)(wrow + (p < 4 ? 2 * p : 16 + 2 * (p - 4)));
        B[2 * p]     = (__bf16)v.x;
        B[2 * p + 1] = (__bf16)v.y;
      }
      acc = __builtin_amdgcn_wmma_f32_16x16x32_bf16(
          false, A[ks], false, B, (short)0, acc, false, false);
    }
    const float bj = fcb[j];
    const float fj = fus[j];
#pragma unroll
    for (int g = 0; g < 8; ++g) partial += fj * tanhf(acc[g] + bj);
  }
  partial = wave_reduce_sum(partial);
  if (lane == 0) atomicAdd(beta_accum, partial);
}

// ---------------------------------------------------------------------------
// Kernel C1: global softmax partials.  One wave per row (wave-stride),
// online (m, S, V) accumulation; exactly GPARTS waves launched.
// ---------------------------------------------------------------------------
__global__ void glob_partial_kernel(const float* __restrict__ h0,
                                    const float* __restrict__ h1,
                                    const float* __restrict__ h2,
                                    const float* __restrict__ gvec,
                                    float* __restrict__ pm,
                                    float* __restrict__ ps,
                                    float* __restrict__ pv,   // GPARTS x 64
                                    int n0, int n1, int n2) {
  const int wave = threadIdx.x >> 5;
  const int lane = threadIdx.x & 31;
  const int wid  = blockIdx.x * (blockDim.x >> 5) + wave;
  const int nw   = gridDim.x * (blockDim.x >> 5);
  const float2 g2 = *(const float2*)(gvec + 2 * lane);
  const int total = n0 + n1 + n2;

  float m = -INFINITY, S = 0.f;
  float2 V = {0.f, 0.f};
  for (int r = wid; r < total; r += nw) {
    const float* row;
    if (r < n0)            row = h0 + (size_t)r * 64;
    else if (r < n0 + n1)  row = h1 + (size_t)(r - n0) * 64;
    else                   row = h2 + (size_t)(r - n0 - n1) * 64;
    const float2 h = *(const float2*)(row + 2 * lane);
    const float s = wave_reduce_sum(g2.x * h.x + g2.y * h.y) * 0.125f; // /sqrt(64)
    if (s > m) {
      const float c = __expf(m - s);
      S *= c; V.x *= c; V.y *= c; m = s;
    }
    const float w = __expf(s - m);
    S += w; V.x += w * h.x; V.y += w * h.y;
  }
  if (lane == 0) { pm[wid] = m; ps[wid] = S; }
  *(float2*)(pv + (size_t)wid * 64 + 2 * lane) = V;
}

// Kernel C2: combine GPARTS partials -> global_embed[64] (single block)
__global__ void glob_combine_kernel(const float* __restrict__ pm,
                                    const float* __restrict__ ps,
                                    const float* __restrict__ pv,
                                    float* __restrict__ ge) {
  __shared__ float sm[256];
  __shared__ float scale[GPARTS];
  const int t = threadIdx.x;
  float m = -INFINITY;
  for (int i = t; i < GPARTS; i += 256) m = fmaxf(m, pm[i]);
  sm[t] = m; __syncthreads();
  for (int off = 128; off >= 1; off >>= 1) {
    if (t < off) sm[t] = fmaxf(sm[t], sm[t + off]);
    __syncthreads();
  }
  const float mg = sm[0];
  __syncthreads();
  float Sl = 0.f;
  for (int i = t; i < GPARTS; i += 256) {
    const float c = __expf(pm[i] - mg);
    scale[i] = c;
    Sl += ps[i] * c;
  }
  sm[t] = Sl; __syncthreads();
  for (int off = 128; off >= 1; off >>= 1) {
    if (t < off) sm[t] += sm[t + off];
    __syncthreads();
  }
  const float Sg = sm[0];
  if (t < 64) {
    float acc = 0.f;
    for (int i = 0; i < GPARTS; ++i) acc += pv[(size_t)i * 64 + t] * scale[i];
    ge[t] = acc / Sg;
  }
}

// ---------------------------------------------------------------------------
// Kernel D: out = sigmoid(gate)*(b0*e0 + b1*e1) + (1-sigmoid(gate))*ge
// ---------------------------------------------------------------------------
__global__ void combine_kernel(const float* __restrict__ e0,
                               const float* __restrict__ e1,
                               const float* __restrict__ ge,
                               const float* __restrict__ betaRaw,
                               const float* __restrict__ gate,
                               float* __restrict__ out, int nRows) {
  const size_t i = (size_t)blockIdx.x * blockDim.x + threadIdx.x;  // float2 units
  const size_t totalPairs = (size_t)nRows * 32;
  if (i >= totalPairs) return;
  const float invN = 1.f / (float)nRows;
  const float x0 = betaRaw[0] * invN, x1 = betaRaw[1] * invN;
  const float mx = fmaxf(x0, x1);
  const float ex0 = __expf(x0 - mx), ex1 = __expf(x1 - mx);
  const float den = ex0 + ex1;
  const float b0 = ex0 / den, b1 = ex1 / den;
  const float gv = 1.f / (1.f + __expf(-gate[0]));
  const int d2 = (int)(i & 31);
  const float2 a = ((const float2*)e0)[i];
  const float2 b = ((const float2*)e1)[i];
  const float2 g = ((const float2*)ge)[d2];
  float2 o;
  o.x = gv * (b0 * a.x + b1 * b.x) + (1.f - gv) * g.x;
  o.y = gv * (b0 * a.y + b1 * b.y) + (1.f - gv) * g.y;
  ((float2*)out)[i] = o;
}

// ---------------------------------------------------------------------------
extern "C" void kernel_launch(void* const* d_in, const int* in_sizes, int n_in,
                              void* d_out, int out_size, void* d_ws, size_t ws_size,
                              hipStream_t stream) {
  const float* h0    = (const float*)d_in[0];
  const float* h1    = (const float*)d_in[1];
  const float* h2    = (const float*)d_in[2];
  const int*   nei0  = (const int*)d_in[3];
  const int*   nei1  = (const int*)d_in[4];
  const float* attw0 = (const float*)d_in[5];
  const float* attw1 = (const float*)d_in[6];
  const float* fcw   = (const float*)d_in[7];
  const float* fcb   = (const float*)d_in[8];
  const float* fus   = (const float*)d_in[9];
  const float* glob  = (const float*)d_in[10];
  const float* gate  = (const float*)d_in[11];

  const int N  = in_sizes[0] / 64;
  const int M1 = in_sizes[1] / 64;
  const int M2 = in_sizes[2] / 64;
  const int S0 = in_sizes[3] / N;
  const int S1 = in_sizes[4] / N;
  (void)M1; (void)M2; (void)n_in; (void)out_size; (void)ws_size;

  // workspace layout (floats)
  float* ws      = (float*)d_ws;
  float* e0      = ws;
  float* e1      = e0 + (size_t)N * 64;
  float* betaRaw = e1 + (size_t)N * 64;                 // 2
  float* pm      = betaRaw + 2;                         // GPARTS
  float* ps      = pm + GPARTS;                         // GPARTS
  float* pv      = ps + GPARTS;                         // GPARTS*64
  float* ge      = pv + (size_t)GPARTS * 64;            // 64

  hipMemsetAsync(betaRaw, 0, 2 * sizeof(float), stream);

  const dim3 blk(256);
  const dim3 gridA((N + 7) / 8);
  // neighbor attention (one wave per node)
  if (S0 == 10)      nei_att_kernel<10><<<gridA, blk, 0, stream>>>(h0, h1, nei0, attw0, e0, N);
  else if (S0 == 20) nei_att_kernel<20><<<gridA, blk, 0, stream>>>(h0, h1, nei0, attw0, e0, N);
  else               nei_att_dyn_kernel<<<gridA, blk, 0, stream>>>(h0, h1, nei0, attw0, e0, N, S0);
  if (S1 == 20)      nei_att_kernel<20><<<gridA, blk, 0, stream>>>(h0, h2, nei1, attw1, e1, N);
  else if (S1 == 10) nei_att_kernel<10><<<gridA, blk, 0, stream>>>(h0, h2, nei1, attw1, e1, N);
  else               nei_att_dyn_kernel<<<gridA, blk, 0, stream>>>(h0, h2, nei1, attw1, e1, N, S1);

  // beta partials via WMMA (one wave per 16-row tile, 8 tiles per block)
  const int tiles = (N + 15) / 16;
  const dim3 gridB((tiles + 7) / 8);
  beta_partial_kernel<<<gridB, blk, 0, stream>>>(e0, fcw, fcb, fus, betaRaw + 0, N);
  beta_partial_kernel<<<gridB, blk, 0, stream>>>(e1, fcw, fcb, fus, betaRaw + 1, N);

  // global attention
  glob_partial_kernel<<<dim3(GBLOCKS), blk, 0, stream>>>(h0, h1, h2, glob, pm, ps, pv,
                                                         N, M1, M2);
  glob_combine_kernel<<<dim3(1), blk, 0, stream>>>(pm, ps, pv, ge);

  // final blend
  const size_t pairs = (size_t)N * 32;
  combine_kernel<<<dim3((unsigned)((pairs + 255) / 256)), blk, 0, stream>>>(
      e0, e1, ge, betaRaw, gate, (float*)d_out, N);
}
